// GraphAttention_81527069212880
// MI455X (gfx1250) — matrix operand order repaired and checked by reading, not compiled, not run on hardware
//
#include <hip/hip_runtime.h>

typedef __attribute__((ext_vector_type(16))) _Float16 v16h;
typedef __attribute__((ext_vector_type(8)))  _Float16 v8h;
typedef __attribute__((ext_vector_type(2)))  _Float16 v2h;
typedef __attribute__((ext_vector_type(8)))  float    v8f;
typedef __attribute__((ext_vector_type(4)))  float    v4f;

#define NN    16
#define FIN   64
#define NH    4
#define FO    32
#define COUT  128   // NH*FO
#define WAVES 8

// dynamic LDS layout (bytes)
#define OFF_WHT   0                       // _Float16 WhT[128][64]  (col-major W)
#define OFF_A2T   16384                   // _Float16 A2T[16][128]  (row-major A2^T)
#define OFF_ADJ   20480                   // float[256]
#define OFF_WAVE  21504
#define NF_BYTES  4096                    // _Float16 nf[16][128] (row-major)
#define AT_BYTES  2048                    // _Float16 attn[4][16][16]
#define SV_BYTES  512                     // float s[128]
#define SHMEM_TOTAL (OFF_WAVE + WAVES*(NF_BYTES + AT_BYTES + SV_BYTES))

union pk_u { v2h h2; unsigned u; };

__global__ void gat_wmma_kernel(const float* __restrict__ x,
                                const float* __restrict__ W,
                                const float* __restrict__ attv,
                                const float* __restrict__ adj,
                                float* __restrict__ out,
                                int nBatch) {
  extern __shared__ v16h smem_raw[];          // 32B-aligned base
  char* smem = (char*)smem_raw;
  _Float16* WhT   = (_Float16*)(smem + OFF_WHT);
  _Float16* A2T   = (_Float16*)(smem + OFF_A2T);
  float*    adj_s = (float*)(smem + OFF_ADJ);

  const int tid  = threadIdx.x;
  const int wave = tid >> 5;
  const int lane = tid & 31;

  _Float16* nf16 = (_Float16*)(smem + OFF_WAVE + wave * NF_BYTES);
  _Float16* at16 = (_Float16*)(smem + OFF_WAVE + WAVES * NF_BYTES + wave * AT_BYTES);
  float*    sv   = (float*)   (smem + OFF_WAVE + WAVES * (NF_BYTES + AT_BYTES) + wave * SV_BYTES);

  // ---- block staging: W^T (f16), A2^T (f16, from att_vec), adj -------------
  for (int i = tid; i < FIN * COUT; i += 256) {
    int k = i >> 7;                     // W row 0..63
    int c = i & 127;                    // W col 0..127
    WhT[c * FIN + k] = (_Float16)W[i];
  }
  // A2 (128x16): col n<4 -> a1[h=n]; col 4..7 -> a2[h=n-4]; else 0.
  for (int i = tid; i < 16 * COUT; i += 256) {
    int n = i >> 7;                     // A2 column
    int k = i & 127;                    // A2 row (nf feature index)
    int h = k >> 5, f = k & 31;
    float v = 0.f;
    if (n < 4 && h == n)                 v = attv[n * 64 + f];
    else if (n >= 4 && n < 8 && h == n - 4) v = attv[(n - 4) * 64 + 32 + f];
    A2T[n * COUT + k] = (_Float16)v;    // row-major A2^T
  }
  adj_s[tid] = adj[tid];
  __syncthreads();

  const int b = blockIdx.x * WAVES + wave;    // wave-uniform
  if (b >= nBatch) return;

  const int m = lane & 15;      // M-row / N-col index inside a 16-tile
  const int g = lane >> 4;      // half-wave group

  // ================= GEMM1: nf = x @ W  (16x64 @ 64x128) ===================
  // A-frag (16x32 f16): elems 0..7 <-> K=8g+e, elems 8..15 <-> K=16+8g+e.
  const float* xr = x + ((size_t)b * NN + m) * FIN + 8 * g;
  v4f q[8];
#pragma unroll
  for (int i = 0; i < 4; ++i) {
    q[2 * i]     = *(const v4f*)(xr + 16 * i);
    q[2 * i + 1] = *(const v4f*)(xr + 16 * i + 4);
  }
  v16h a0, a1;
#pragma unroll
  for (int e = 0; e < 8; ++e) {
    a0[e]     = (_Float16)q[(e >> 2)][e & 3];       // K = 8g + e
    a0[8 + e] = (_Float16)q[2 + (e >> 2)][e & 3];   // K = 16 + 8g + e
    a1[e]     = (_Float16)q[4 + (e >> 2)][e & 3];   // K = 32 + 8g + e
    a1[8 + e] = (_Float16)q[6 + (e >> 2)][e & 3];   // K = 48 + 8g + e
  }

  // bfreg[t]: f16 B-fragment of nf columns 16t..16t+15 for GEMM2
  // (elems 0..7 <-> K=8g+e = row 8g+e of nf; elems 8..15: zero padding)
  v16h bfreg[8];
#pragma unroll
  for (int t = 0; t < 8; ++t) {
    const _Float16* colp = WhT + (16 * t + m) * FIN;
    v8h b0l = *(const v8h*)(colp + 8 * g);
    v8h b0h = *(const v8h*)(colp + 16 + 8 * g);
    v8h b1l = *(const v8h*)(colp + 32 + 8 * g);
    v8h b1h = *(const v8h*)(colp + 48 + 8 * g);
    v16h b0 = __builtin_shufflevector(b0l, b0h, 0,1,2,3,4,5,6,7,8,9,10,11,12,13,14,15);  // K chunk 0
    v16h b1 = __builtin_shufflevector(b1l, b1h, 0,1,2,3,4,5,6,7,8,9,10,11,12,13,14,15);  // K chunk 1
    v8f acc = {};
    acc = __builtin_amdgcn_wmma_f32_16x16x32_f16(false, a0, false, b0,
                                                 (short)0, acc, false, false);
    acc = __builtin_amdgcn_wmma_f32_16x16x32_f16(false, a1, false, b1,
                                                 (short)0, acc, false, false);
    // pack row pairs to f16 once; reuse for LDS store + register B-frag
    pk_u pk[4];
#pragma unroll
    for (int j = 0; j < 4; ++j) {
      v2h p;
      p[0] = (_Float16)acc[2 * j];
      p[1] = (_Float16)acc[2 * j + 1];
      pk[j].h2 = p;
    }
    // nf16 row-major (rows r+8g, col 16t+m) -- feeds S-WMMA B-fragments
#pragma unroll
    for (int r = 0; r < 8; ++r)
      nf16[(r + 8 * g) * COUT + 16 * t + m] = pk[r >> 1].h2[r & 1];
    // register B-frag: own rows (0..7 on g=0) + zero K padding
    union { v16h h; unsigned u[8]; } bfu;
#pragma unroll
    for (int j = 0; j < 4; ++j) {
      unsigned d  = pk[j].u;
      unsigned up = (unsigned)__shfl_xor((int)d, 16, 32);
      bfu.u[j]     = d;                    // K = 8g+2j, 8g+2j+1 (own rows)
      bfu.u[4 + j] = 0u;                   // K padding
    }
    bfreg[t] = bfu.h;
  }

  // ================= S^T = A2^T @ nf^T via WMMA ============================
  // B-frag of nf^T: lane=col i (nf row), elems 0..7 <-> K=32ck+8g+e, 8..15 <-> K=32ck+16+8g+(e-8) (nf col)
  //   -> two v8h row-loads from row-major nf16.
  v8f sacc = {};
#pragma unroll
  for (int ck = 0; ck < 4; ++ck) {
    const _Float16* ap = A2T + m * COUT + 32 * ck;  // row m of A2^T
    v8h lo = *(const v8h*)(ap + 8 * g);             // K = 32ck + 8g + e
    v8h hi = *(const v8h*)(ap + 16 + 8 * g);        // K = 32ck + 16 + 8g + e
    v16h aS;
#pragma unroll
    for (int e = 0; e < 8; ++e) { aS[e] = lo[e]; aS[8 + e] = hi[e]; }
    v8h bSl = *(const v8h*)(nf16 + m * COUT + 32 * ck + 8 * g);
    v8h bSh = *(const v8h*)(nf16 + m * COUT + 32 * ck + 16 + 8 * g);
    v16h bS = __builtin_shufflevector(bSl, bSh, 0,1,2,3,4,5,6,7,8,9,10,11,12,13,14,15);
    sacc = __builtin_amdgcn_wmma_f32_16x16x32_f16(false, aS, false, bS,
                                                  (short)0, sacc, false, false);
  }
  // D: lane col = node i = m; row n = r+8g.  n<4 -> s1[i][n], 4..7 -> s2.
  if (g == 0) {
#pragma unroll
    for (int r = 0; r < 8; ++r) sv[r * 16 + m] = sacc[r];
  }

  // ================= leaky-relu + mask + softmax -> attn (f16) =============
#pragma unroll
  for (int p = 0; p < 2; ++p) {
    int pr = lane + 32 * p;          // (node i, head h)
    int i  = pr & 15;
    int h  = pr >> 4;
    float s1 = sv[h * 16 + i];
    float e[16];
    float mx = -3.0e38f;
#pragma unroll
    for (int j = 0; j < 16; ++j) {
      float sc = s1 + sv[64 + h * 16 + j];
      sc = (sc > 0.f) ? sc : 0.2f * sc;
      sc = (adj_s[i * 16 + j] != 0.f) ? sc : -3.0e38f;
      e[j] = sc;
      mx = fmaxf(mx, sc);
    }
    float sum = 0.f;
#pragma unroll
    for (int j = 0; j < 16; ++j) {
      float ev = (e[j] > -1.0e38f) ? __expf(e[j] - mx) : 0.f;
      e[j] = ev;
      sum += ev;
    }
    float inv = 1.f / sum;
    v16h arow;
#pragma unroll
    for (int j = 0; j < 16; ++j) arow[j] = (_Float16)(e[j] * inv);
    *(v16h*)(at16 + h * 256 + i * 16) = arow;   // one 32B LDS store
  }

  // ================= GEMM2: out_h = attn_h @ nf_h (B from registers) =======
#pragma unroll
  for (int h = 0; h < NH; ++h) {
    // A-frag: attn row m; elems 0..7 = K=8g+e (real), 8..15 = 0 (K padding)
    v8h arow8 = *(const v8h*)(at16 + h * 256 + m * 16 + 8 * g);
    v16h af = {};
#pragma unroll
    for (int e = 0; e < 8; ++e) af[e] = arow8[e];
#pragma unroll
    for (int t = 0; t < 2; ++t) {
      v8f oacc = {};
      oacc = __builtin_amdgcn_wmma_f32_16x16x32_f16(false, af, false,
                                                    bfreg[2 * h + t],
                                                    (short)0, oacc, false, false);
      float* orow = out + (size_t)b * NN * COUT + 32 * h + 16 * t + m;
#pragma unroll
      for (int r = 0; r < 8; ++r)
        orow[(size_t)(r + 8 * g) * COUT] = oacc[r];
    }
  }
}

extern "C" void kernel_launch(void* const* d_in, const int* in_sizes, int n_in,
                              void* d_out, int out_size, void* d_ws, size_t ws_size,
                              hipStream_t stream) {
  const float* x    = (const float*)d_in[0];
  const float* W    = (const float*)d_in[1];
  const float* attv = (const float*)d_in[2];
  const float* adj  = (const float*)d_in[3];
  float* out = (float*)d_out;

  int nB = in_sizes[0] / (NN * FIN);
  int blocks = (nB + WAVES - 1) / WAVES;
  gat_wmma_kernel<<<blocks, 256, SHMEM_TOTAL, stream>>>(x, W, attv, adj, out, nB);
}
